// NaMMAttention_21423296872967
// MI455X (gfx1250) — compile-verified, matrix operand order checked by reading, not run on hardware
//
#include <hip/hip_runtime.h>
#include <hip/hip_bf16.h>
#include <cstdint>

// ---------------------------------------------------------------------------
// Problem constants (match reference)
// ---------------------------------------------------------------------------
#define BB     2
#define LV     2048
#define LT     128
#define LTOT   2176          // LV + LT
#define HH_    16
#define DD     128
#define INNER  2048          // H*D
#define NQKV   6144          // 3*INNER
#define DIM    2048          // VID_DIM == TXT_DIM

#define LOG2E      1.4426950408889634f
#define ATT_SCALE  0.08838834764831845f   // D^-0.5
#define LOG2_10000 13.287712379549449f

typedef __attribute__((ext_vector_type(16))) __bf16        v16bf;
typedef __attribute__((ext_vector_type(8)))  float         v8f;
typedef __attribute__((ext_vector_type(4)))  unsigned int  v4u;
typedef __attribute__((ext_vector_type(8)))  int           v8i;
typedef __attribute__((ext_vector_type(4)))  int           v4i;

union Frag {            // one 16x32 (A) or 32x16-N-major (B) bf16 fragment: 8 VGPRs
    v16bf v;
    uint4 q[2];
};

static __device__ __forceinline__ v8f wmma_bf16(const v16bf& a, const v16bf& b, v8f c) {
    return __builtin_amdgcn_wmma_f32_16x16x32_bf16(
        /*neg_a=*/false, a, /*neg_b=*/false, b,
        /*c_mod=*/(short)0, c, /*reuse_a=*/false, /*reuse_b=*/false);
}

// ---------------------------------------------------------------------------
// Tensor Data Mover: 2D tile load (global -> LDS), bf16 elements.
// D# packing per CDNA5 ISA.
// This toolchain exposes the 6-arg builtin:
//   (uint32x4 g0, int32x8 g1, int32x4 g2, int32x4 g3, int32x8 g4, i32 cpol)
//   tile_d0  : elements per row (dim0), multiple of 4B total
//   tile_d1  : number of rows
//   stride_d0: tensor dim0 stride in elements
//   pad: interval code 3 = 16 DWORDs (64B = one 32-elem bf16 row),
//        amount  code 3 = 4 DWORDs  (16B = 8 elems) -> LDS row stride 40 elems
// ---------------------------------------------------------------------------
static __device__ __forceinline__ void tdm_load_2d_bf16(
    const void* gaddr, unsigned lds_off,
    unsigned tensor_d0, unsigned tensor_d1,
    unsigned tile_d0, unsigned tile_d1,
    unsigned stride_d0,
    bool pad_en, unsigned pad_interval_code, unsigned pad_amount_code)
{
    unsigned long long ga = (unsigned long long)(uintptr_t)gaddr;
    v4u g0;
    g0[0] = 1u;                                              // count=1, user mode
    g0[1] = lds_off;                                         // lds_addr
    g0[2] = (unsigned)(ga & 0xffffffffu);                    // global_addr lo
    g0[3] = (unsigned)((ga >> 32) & 0x01ffffffu) | (2u << 30); // ga hi | type=2
    v8i g1;
    unsigned w0 = (1u << 16);                                // data_size=1 (2 bytes)
    if (pad_en)
        w0 |= (1u << 20) | (pad_interval_code << 22) | (pad_amount_code << 25);
    g1[0] = (int)w0;                                         // wg_mask=0, flags
    g1[1] = (int)((tensor_d0 & 0xffffu) << 16);              // tensor_dim0 lo16
    g1[2] = (int)(((tensor_d0 >> 16) & 0xffffu) |
                  ((tensor_d1 & 0xffffu) << 16));            // d0 hi | d1 lo
    g1[3] = (int)(((tensor_d1 >> 16) & 0xffffu) |
                  ((tile_d0 & 0xffffu) << 16));              // d1 hi | tile_dim0
    g1[4] = (int)(tile_d1 & 0xffffu);                        // tile_dim1 | tile_dim2=0
    g1[5] = (int)stride_d0;                                  // dim0_stride lo32
    g1[6] = 0;                                               // stride hi | dim1_stride lo
    g1[7] = 0;
    v4i g2 = {0, 0, 0, 0};
    v4i g3 = {0, 0, 0, 0};
    v8i g4 = {0, 0, 0, 0, 0, 0, 0, 0};
    __builtin_amdgcn_tensor_load_to_lds(g0, g1, g2, g3, g4, 0);
}

// ---------------------------------------------------------------------------
// 1) fp32 -> bf16 cast (grid-stride)
// ---------------------------------------------------------------------------
__global__ __launch_bounds__(256) void cast_f32_bf16(const float* __restrict__ src,
                                                     __bf16* __restrict__ dst,
                                                     size_t n) {
    size_t i = (size_t)blockIdx.x * blockDim.x + threadIdx.x;
    size_t stride = (size_t)gridDim.x * blockDim.x;
    for (; i < n; i += stride) dst[i] = (__bf16)src[i];
}

// ---------------------------------------------------------------------------
// 2) bf16 GEMM: C(f32, MxN) = A(bf16, MxK rm) * B(bf16, KxN rm) [+ bias]
//    128x128 tile per block (256 threads = 8 waves), BK=32.
//    A tile staged by the Tensor Data Mover (wave 0, overlapped with the
//    VGPR-transpose staging of B by all waves). Each wave: 64x32 of C.
// ---------------------------------------------------------------------------
#define GBM 128
#define GBN 128
#define GBK 32
#define GLD 40            // padded LDS row stride (elements): 80B, 16B aligned

__global__ __launch_bounds__(256) void gemm_bf16(const __bf16* __restrict__ A,
                                                 const __bf16* __restrict__ Bm,
                                                 float* __restrict__ C,
                                                 const float* __restrict__ bias,
                                                 int M, int N, int K) {
    __shared__ __bf16 sA[GBM * GLD];     // row-major A tile (TDM-padded rows)
    __shared__ __bf16 sB[GBN * GLD];     // N-major (transposed) B tile

    const int tid  = threadIdx.x;
    const int lane = tid & 31;
    const int wave = tid >> 5;
    const int wm   = wave >> 2;          // 0..1  -> 64-row slab
    const int wn   = wave & 3;           // 0..3  -> 32-col slab
    const int l16  = lane & 15;
    const int hseg = lane >> 4;          // 0/1: which K-halves this lane holds
    const int m0   = blockIdx.y * GBM;
    const int n0   = blockIdx.x * GBN;
    const unsigned sA_off = (unsigned)(uintptr_t)(&sA[0]);   // LDS byte offset

    v8f acc[4][2];
#pragma unroll
    for (int i = 0; i < 4; ++i)
#pragma unroll
        for (int j = 0; j < 2; ++j) acc[i][j] = (v8f){0.f,0.f,0.f,0.f,0.f,0.f,0.f,0.f};

    for (int k0 = 0; k0 < K; k0 += GBK) {
        // ---- A tile via Tensor Data Mover: 128 rows x 32 bf16, padded -----
        if (wave == 0) {
            tdm_load_2d_bf16(A + (size_t)m0 * K + k0, sA_off,
                             /*tensor_d0=*/(unsigned)(K - k0),
                             /*tensor_d1=*/(unsigned)(M - m0),
                             /*tile_d0=*/GBK, /*tile_d1=*/GBM,
                             /*stride_d0=*/(unsigned)K,
                             /*pad=*/true, /*interval 16dw*/3, /*amount 4dw*/3);
        }
        // ---- stage B tile transposed: sB[n][k] (overlaps the TDM) ---------
        {
            int idx0 = tid * 2, idx1 = tid * 2 + 1;          // 0..511
            uint4 val0 = *(const uint4*)(Bm + (size_t)(k0 + (idx0 >> 4)) * N +
                                         n0 + (idx0 & 15) * 8);
            uint4 val1 = *(const uint4*)(Bm + (size_t)(k0 + (idx1 >> 4)) * N +
                                         n0 + (idx1 & 15) * 8);
            const unsigned short* u0 = (const unsigned short*)&val0;
            const unsigned short* u1 = (const unsigned short*)&val1;
#pragma unroll
            for (int e = 0; e < 8; ++e)
                ((unsigned short*)sB)[((idx0 & 15) * 8 + e) * GLD + (idx0 >> 4)] = u0[e];
#pragma unroll
            for (int e = 0; e < 8; ++e)
                ((unsigned short*)sB)[((idx1 & 15) * 8 + e) * GLD + (idx1 >> 4)] = u1[e];
        }
        // ---- prefetch next B tile into cache (global_prefetch_b8) ---------
        if (k0 + GBK < K)
            __builtin_prefetch(Bm + (size_t)(k0 + GBK + (tid >> 3)) * N + n0, 0, 1);

        if (wave == 0) __builtin_amdgcn_s_wait_tensorcnt(0); // drain TDM
        __syncthreads();

        // ---- fragments: two b128 LDS loads each (ISA 16-bit A/B layout) ---
        Frag fa[4];
#pragma unroll
        for (int i = 0; i < 4; ++i) {
            const __bf16* p = &sA[(wm * 64 + i * 16 + l16) * GLD];
            fa[i].q[0] = *(const uint4*)(p + hseg * 8);
            fa[i].q[1] = *(const uint4*)(p + 16 + hseg * 8);
        }
        Frag fb[2];
#pragma unroll
        for (int j = 0; j < 2; ++j) {
            const __bf16* p = &sB[(wn * 32 + j * 16 + l16) * GLD];
            fb[j].q[0] = *(const uint4*)(p + hseg * 8);
            fb[j].q[1] = *(const uint4*)(p + 16 + hseg * 8);
        }
#pragma unroll
        for (int i = 0; i < 4; ++i)
#pragma unroll
            for (int j = 0; j < 2; ++j)
                acc[i][j] = wmma_bf16(fa[i].v, fb[j].v, acc[i][j]);
        __syncthreads();
    }

    // ---- epilogue: C-tile layout lane<16 -> M=r, lane>=16 -> M=8+r --------
#pragma unroll
    for (int i = 0; i < 4; ++i)
#pragma unroll
        for (int j = 0; j < 2; ++j)
#pragma unroll
            for (int r = 0; r < 8; ++r) {
                int mr = m0 + wm * 64 + i * 16 + hseg * 8 + r;
                int nc = n0 + wn * 32 + j * 16 + l16;
                float v = acc[i][j][r];
                if (bias) v += bias[nc];
                if (mr < M && nc < N) C[(size_t)mr * N + nc] = v;
            }
}

// ---------------------------------------------------------------------------
// 3) qkv_post: RMSNorm(q,k) + RoPE(vid q,k) + cast, write Q/K/V (B,H,L,D) bf16
//    one wave per (b, l, h) row; D=128 -> 4 elems/lane
// ---------------------------------------------------------------------------
__global__ __launch_bounds__(256) void qkv_post(const float* __restrict__ qkvV,
                                                const float* __restrict__ qkvT,
                                                const float* __restrict__ gqv,
                                                const float* __restrict__ gqt,
                                                const float* __restrict__ gkv,
                                                const float* __restrict__ gkt,
                                                __bf16* __restrict__ Qb,
                                                __bf16* __restrict__ Kb,
                                                __bf16* __restrict__ Vb) {
    const int wave = threadIdx.x >> 5, lane = threadIdx.x & 31;
    int gid = blockIdx.x * 8 + wave;            // (b, l, h) flattened
    int h = gid % HH_;
    int t = gid / HH_;
    int l = t % LTOT;
    int b = t / LTOT;
    const bool isVid = (l < LV);

    const float* src = isVid ? qkvV + (size_t)(b * LV + l) * NQKV
                             : qkvT + (size_t)(b * LT + (l - LV)) * NQKV;
    const float* gq = isVid ? gqv : gqt;
    const float* gk = isVid ? gkv : gkt;

    float q[4], k[4], v[4];
    float sq = 0.f, sk = 0.f;
#pragma unroll
    for (int i = 0; i < 4; ++i) {
        int d = lane + 32 * i;
        q[i] = src[(size_t)h * DD + d];                      // c = 0
        k[i] = src[(size_t)(HH_ + h) * DD + d];              // c = 1
        v[i] = src[(size_t)(2 * HH_ + h) * DD + d];          // c = 2
        sq += q[i] * q[i];
        sk += k[i] * k[i];
    }
#pragma unroll
    for (int m = 16; m >= 1; m >>= 1) {
        sq += __shfl_xor(sq, m);
        sk += __shfl_xor(sk, m);
    }
    const float rq = rsqrtf(sq * (1.f / DD) + 1e-6f);
    const float rk = rsqrtf(sk * (1.f / DD) + 1e-6f);
#pragma unroll
    for (int i = 0; i < 4; ++i) {
        int d = lane + 32 * i;
        q[i] *= rq * gq[d];
        k[i] *= rk * gk[d];
    }
    if (isVid) {   // RoPE: pairs (d, d+64) live in the same lane (i, i+2)
#pragma unroll
        for (int i = 0; i < 2; ++i) {
            int j = lane + 32 * i;                           // 0..63
            float inv = exp2f(-(float)j * (LOG2_10000 / 64.f));
            float ang = (float)l * inv;
            float s, c;
            __sincosf(ang, &s, &c);
            float q1 = q[i], q2 = q[i + 2];
            q[i]     = q1 * c - q2 * s;
            q[i + 2] = q1 * s + q2 * c;
            float k1 = k[i], k2 = k[i + 2];
            k[i]     = k1 * c - k2 * s;
            k[i + 2] = k1 * s + k2 * c;
        }
    }
    const size_t ob = ((size_t)(b * HH_ + h) * LTOT + l) * DD;
#pragma unroll
    for (int i = 0; i < 4; ++i) {
        int d = lane + 32 * i;
        Qb[ob + d] = (__bf16)q[i];
        Kb[ob + d] = (__bf16)k[i];
        Vb[ob + d] = (__bf16)v[i];
    }
}

// ---------------------------------------------------------------------------
// 4) flash attention: block = (b,h) x 8 q-tiles; wave owns one 16-row Q tile
// ---------------------------------------------------------------------------
#define ALDV 40
#define ALDP 40

__global__ __launch_bounds__(256) void attn_fwd(const __bf16* __restrict__ Qg,
                                                const __bf16* __restrict__ Kg,
                                                const __bf16* __restrict__ Vg,
                                                __bf16* __restrict__ Og) {
    __shared__ __bf16 sVT[DD * ALDV];           // V chunk transposed: [d][k]
    __shared__ __bf16 sP[8 * 16 * ALDP];        // per-wave P tiles [16][32]

    const int tid  = threadIdx.x;
    const int lane = tid & 31;
    const int wave = tid >> 5;
    const int l16  = lane & 15;
    const int hseg = lane >> 4;
    const int b = blockIdx.x / HH_, h = blockIdx.x % HH_;
    const int q0 = (blockIdx.y * 8 + wave) * 16;          // 17*8 = 136 = LTOT/16 exactly

    const __bf16* Qp = Qg + (size_t)(b * HH_ + h) * LTOT * DD;
    const __bf16* Kp = Kg + (size_t)(b * HH_ + h) * LTOT * DD;
    const __bf16* Vp = Vg + (size_t)(b * HH_ + h) * LTOT * DD;
    __bf16* Pw = sP + wave * 16 * ALDP;

    // Q fragments (A layout), straight from global
    Frag aq[4];
    const __bf16* qrow = Qp + (size_t)(q0 + l16) * DD;
#pragma unroll
    for (int c = 0; c < 4; ++c) {
        aq[c].q[0] = *(const uint4*)(qrow + c * 32 + hseg * 8);
        aq[c].q[1] = *(const uint4*)(qrow + c * 32 + 16 + hseg * 8);
    }

    float m[8], lsum[8];
    v8f o[8];
#pragma unroll
    for (int r = 0; r < 8; ++r) { m[r] = -1e30f; lsum[r] = 0.f; }
#pragma unroll
    for (int dc = 0; dc < 8; ++dc) o[dc] = (v8f){0.f,0.f,0.f,0.f,0.f,0.f,0.f,0.f};

    for (int kc = 0; kc < LTOT; kc += 32) {
        // ---- cooperative stage of V^T chunk: sVT[d][k], k in [0,32) -------
        {
            int idx0 = tid * 2, idx1 = tid * 2 + 1;          // 0..511
            uint4 val0 = *(const uint4*)(Vp + (size_t)(kc + (idx0 >> 4)) * DD +
                                         (idx0 & 15) * 8);
            uint4 val1 = *(const uint4*)(Vp + (size_t)(kc + (idx1 >> 4)) * DD +
                                         (idx1 & 15) * 8);
            const unsigned short* u0 = (const unsigned short*)&val0;
            const unsigned short* u1 = (const unsigned short*)&val1;
#pragma unroll
            for (int e = 0; e < 8; ++e)
                ((unsigned short*)sVT)[((idx0 & 15) * 8 + e) * ALDV + (idx0 >> 4)] = u0[e];
#pragma unroll
            for (int e = 0; e < 8; ++e)
                ((unsigned short*)sVT)[((idx1 & 15) * 8 + e) * ALDV + (idx1 >> 4)] = u1[e];
        }
        __syncthreads();
        __builtin_prefetch(Vp + (size_t)(kc + 32 + (tid >> 3)) * DD, 0, 1);

        // ---- S = Q * K^T : two 16x16 score tiles, K frags from global -----
        v8f s0 = (v8f){0.f,0.f,0.f,0.f,0.f,0.f,0.f,0.f};
        v8f s1 = (v8f){0.f,0.f,0.f,0.f,0.f,0.f,0.f,0.f};
#pragma unroll
        for (int c = 0; c < 4; ++c) {
            Frag bk0, bk1;
            const __bf16* kr0 = Kp + (size_t)(kc + l16) * DD + c * 32;
            const __bf16* kr1 = Kp + (size_t)(kc + 16 + l16) * DD + c * 32;
            bk0.q[0] = *(const uint4*)(kr0 + hseg * 8);
            bk0.q[1] = *(const uint4*)(kr0 + 16 + hseg * 8);
            bk1.q[0] = *(const uint4*)(kr1 + hseg * 8);
            bk1.q[1] = *(const uint4*)(kr1 + 16 + hseg * 8);
            s0 = wmma_bf16(aq[c].v, bk0.v, s0);
            s1 = wmma_bf16(aq[c].v, bk1.v, s1);
        }

        // ---- online softmax (row reductions across 16-lane halves) --------
        float alpha[8];
#pragma unroll
        for (int r = 0; r < 8; ++r) {
            float a0 = s0[r] * ATT_SCALE;
            float a1 = s1[r] * ATT_SCALE;
            float mt = fmaxf(a0, a1);
#pragma unroll
            for (int mk = 8; mk >= 1; mk >>= 1) mt = fmaxf(mt, __shfl_xor(mt, mk));
            float mn = fmaxf(m[r], mt);
            alpha[r] = exp2f((m[r] - mn) * LOG2E);
            float p0 = exp2f((a0 - mn) * LOG2E);
            float p1 = exp2f((a1 - mn) * LOG2E);
            float rs = p0 + p1;
#pragma unroll
            for (int mk = 8; mk >= 1; mk >>= 1) rs += __shfl_xor(rs, mk);
            lsum[r] = lsum[r] * alpha[r] + rs;
            m[r] = mn;
            Pw[(hseg * 8 + r) * ALDP + l16]      = (__bf16)p0;   // C-layout -> LDS
            Pw[(hseg * 8 + r) * ALDP + 16 + l16] = (__bf16)p1;
        }
#pragma unroll
        for (int dc = 0; dc < 8; ++dc)
#pragma unroll
            for (int r = 0; r < 8; ++r) o[dc][r] *= alpha[r];

        // wave-local LDS RAW: P written by all lanes, re-read transposed
        asm volatile("s_wait_dscnt 0x0" ::: "memory");

        // ---- O += P * V ---------------------------------------------------
        Frag pa;
        const __bf16* pr = Pw + l16 * ALDP;
        pa.q[0] = *(const uint4*)(pr + hseg * 8);
        pa.q[1] = *(const uint4*)(pr + 16 + hseg * 8);
#pragma unroll
        for (int dc = 0; dc < 8; ++dc) {
            Frag bv;
            const __bf16* vr = sVT + (dc * 16 + l16) * ALDV;
            bv.q[0] = *(const uint4*)(vr + hseg * 8);
            bv.q[1] = *(const uint4*)(vr + 16 + hseg * 8);
            o[dc] = wmma_bf16(pa.v, bv.v, o[dc]);
        }
        __syncthreads();
    }

    // ---- epilogue: O (B, L, H*D) bf16 ------------------------------------
#pragma unroll
    for (int dc = 0; dc < 8; ++dc)
#pragma unroll
        for (int r = 0; r < 8; ++r) {
            float val = o[dc][r] / lsum[r];
            int row = q0 + hseg * 8 + r;
            Og[((size_t)b * LTOT + row) * INNER + (size_t)h * DD + dc * 16 + l16] =
                (__bf16)val;
        }
}

// ---------------------------------------------------------------------------
// host-side launch
// ---------------------------------------------------------------------------
extern "C" void kernel_launch(void* const* d_in, const int* in_sizes, int n_in,
                              void* d_out, int out_size, void* d_ws, size_t ws_size,
                              hipStream_t stream) {
    const float* vid      = (const float*)d_in[0];
    const float* txt      = (const float*)d_in[1];
    const float* Wqkv_vid = (const float*)d_in[4];
    const float* Wqkv_txt = (const float*)d_in[5];
    const float* gq_vid   = (const float*)d_in[6];
    const float* gq_txt   = (const float*)d_in[7];
    const float* gk_vid   = (const float*)d_in[8];
    const float* gk_txt   = (const float*)d_in[9];
    const float* Wout_vid = (const float*)d_in[10];
    const float* Wout_txt = (const float*)d_in[11];
    const float* bout_vid = (const float*)d_in[12];
    const float* bout_txt = (const float*)d_in[13];
    float* out = (float*)d_out;

    // ---- carve workspace (256B aligned) ----------------------------------
    char* ws = (char*)d_ws;
    auto carve = [&](size_t bytes) {
        char* p = ws;
        ws += (bytes + 255) & ~(size_t)255;
        return p;
    };
    const size_t nVid  = (size_t)BB * LV * DIM;      // 8,388,608
    const size_t nTxt  = (size_t)BB * LT * DIM;      //   524,288
    const size_t nWqkv = (size_t)DIM * NQKV;         // 12,582,912
    const size_t nWout = (size_t)INNER * DIM;        //  4,194,304
    const size_t nQKVv = (size_t)BB * LV * NQKV;
    const size_t nQKVt = (size_t)BB * LT * NQKV;
    const size_t nHeads = (size_t)BB * HH_ * LTOT * DD;
    const size_t nO     = (size_t)BB * LTOT * INNER;

    __bf16* vidB = (__bf16*)carve(nVid * 2);
    __bf16* txtB = (__bf16*)carve(nTxt * 2);
    __bf16* WqvB = (__bf16*)carve(nWqkv * 2);
    __bf16* WqtB = (__bf16*)carve(nWqkv * 2);
    __bf16* WovB = (__bf16*)carve(nWout * 2);
    __bf16* WotB = (__bf16*)carve(nWout * 2);
    float*  qkvV = (float*)carve(nQKVv * 4);
    float*  qkvT = (float*)carve(nQKVt * 4);
    __bf16* Qb   = (__bf16*)carve(nHeads * 2);
    __bf16* Kb   = (__bf16*)carve(nHeads * 2);
    __bf16* Vb   = (__bf16*)carve(nHeads * 2);
    __bf16* Ob   = (__bf16*)carve(nO * 2);

    // ---- 1) casts --------------------------------------------------------
    auto cast = [&](const float* s, __bf16* d, size_t n) {
        int blocks = (int)((n + 1023) / 1024);
        cast_f32_bf16<<<blocks, 256, 0, stream>>>(s, d, n);
    };
    cast(vid, vidB, nVid);
    cast(txt, txtB, nTxt);
    cast(Wqkv_vid, WqvB, nWqkv);
    cast(Wqkv_txt, WqtB, nWqkv);
    cast(Wout_vid, WovB, nWout);
    cast(Wout_txt, WotB, nWout);

    // ---- 2) QKV projections ---------------------------------------------
    {
        dim3 g(NQKV / GBN, (BB * LV) / GBM);
        gemm_bf16<<<g, 256, 0, stream>>>(vidB, WqvB, qkvV, nullptr,
                                         BB * LV, NQKV, DIM);
    }
    {
        dim3 g(NQKV / GBN, (BB * LT) / GBM);
        gemm_bf16<<<g, 256, 0, stream>>>(txtB, WqtB, qkvT, nullptr,
                                         BB * LT, NQKV, DIM);
    }

    // ---- 3) rmsnorm + rope + pack ---------------------------------------
    {
        int blocks = (BB * LTOT * HH_) / 8;     // 8704
        qkv_post<<<blocks, 256, 0, stream>>>(qkvV, qkvT, gq_vid, gq_txt,
                                             gk_vid, gk_txt, Qb, Kb, Vb);
    }

    // ---- 4) attention ----------------------------------------------------
    {
        dim3 g(BB * HH_, LTOT / (16 * 8));      // (32, 17)
        attn_fwd<<<g, 256, 0, stream>>>(Qb, Kb, Vb, Ob);
    }

    // ---- 5) output projections (per batch: O rows are strided by LTOT) ---
    float* out_vid = out;
    float* out_txt = out + (size_t)BB * LV * DIM;
    for (int b = 0; b < BB; ++b) {
        dim3 gv(DIM / GBN, LV / GBM);
        gemm_bf16<<<gv, 256, 0, stream>>>(Ob + (size_t)b * LTOT * INNER, WovB,
                                          out_vid + (size_t)b * LV * DIM,
                                          bout_vid, LV, DIM, INNER);
        dim3 gt(DIM / GBN, LT / GBM);
        gemm_bf16<<<gt, 256, 0, stream>>>(Ob + ((size_t)b * LTOT + LV) * INNER, WotB,
                                          out_txt + (size_t)b * LT * DIM,
                                          bout_txt, LT, DIM, INNER);
    }
}
